// HGIN_classifier_29609504539447
// MI455X (gfx1250) — matrix operand order repaired and checked by reading, not compiled, not run on hardware
//
#include <hip/hip_runtime.h>
#include <hip/hip_bf16.h>

// ---- problem constants (match reference) ----
#define NN   100000
#define EE   1600000
#define RR   2
#define DIN  128
#define DH   256
#define DOUT 2

typedef __attribute__((ext_vector_type(16))) __bf16 v16bf;
typedef __attribute__((ext_vector_type(8)))  float  v8f;

// ---------------------------------------------------------------------------
// Kernel 1: zero the layer-1 aggregation buffer (N*R*DIN floats)
// ---------------------------------------------------------------------------
__global__ __launch_bounds__(256) void k_zero(float4* p, int n4) {
    int i = blockIdx.x * 256 + threadIdx.x;
    if (i < n4) p[i] = make_float4(0.f, 0.f, 0.f, 0.f);
}

// ---------------------------------------------------------------------------
// Kernel 2: layer-1 edge scatter. One 32-lane "virtual wave" per edge,
// each lane moves a float4 of x[src] into agg[dst, etype] via f32 atomics.
// ---------------------------------------------------------------------------
__global__ __launch_bounds__(256) void k_scatter1(const float* __restrict__ x,
                                                  const int* __restrict__ ei,
                                                  const int* __restrict__ et,
                                                  float* __restrict__ agg) {
    int tid  = blockIdx.x * 256 + threadIdx.x;
    int e    = tid >> 5;
    int lane = tid & 31;
    if (e >= EE) return;
    int src = ei[e];
    int dst = ei[EE + e];
    int r   = et[e];
    const float4 v = *(const float4*)(x + (size_t)src * DIN + lane * 4);
    float* a = agg + ((size_t)dst * RR + r) * DIN + lane * 4;
    unsafeAtomicAdd(a + 0, v.x);
    unsafeAtomicAdd(a + 1, v.y);
    unsafeAtomicAdd(a + 2, v.z);
    unsafeAtomicAdd(a + 3, v.w);
}

// ---------------------------------------------------------------------------
// Kernel 3: fused layer-1 GEMM:
//   h = relu( [x | agg0 | agg1] (N x 384) @ [W1s; W1r0; W1r1] (384 x 256) + b1 )
// bf16 WMMA with f32 accumulation. Block = 256 thr = 8 waves -> 128x64 tile.
// 12 K-steps of 32. LDS stride padded to 40 bf16 (80 B) for bank spread.
// ---------------------------------------------------------------------------
#define ASTRIDE 40
__global__ __launch_bounds__(256) void k_gemm1(const float* __restrict__ x,
                                               const float* __restrict__ agg,
                                               const float* __restrict__ w1s,
                                               const float* __restrict__ w1r,
                                               const float* __restrict__ b1,
                                               float* __restrict__ h) {
    __shared__ __bf16 Abf[128][ASTRIDE];   // [row][k]     10,240 B
    __shared__ __bf16 Bbf[64][ASTRIDE];    // [col][k]      5,120 B

    const int tid  = threadIdx.x;
    const int wave = tid >> 5;
    const int lane = tid & 31;
    const int half = lane >> 4;
    const int l16  = lane & 15;

    const int rowBase = (blockIdx.x >> 2) * 128;
    const int colBase = (blockIdx.x & 3) * 64;

    v8f acc[4];
    v8f zero = {};
    #pragma unroll
    for (int t = 0; t < 4; ++t) acc[t] = zero;

    #pragma unroll 1
    for (int s = 0; s < 12; ++s) {
        // ---- stage A: 128 rows x 32 k (f32 -> bf16), 4 float4 per thread ----
        #pragma unroll
        for (int q = 0; q < 4; ++q) {
            int tt    = tid + q * 256;
            int lrow  = tt >> 3;
            int k     = (tt & 7) * 4;
            int rowG  = rowBase + lrow;
            float4 v  = make_float4(0.f, 0.f, 0.f, 0.f);
            if (rowG < NN) {
                const float* src;
                if (s < 4)      src = x   + (size_t)rowG * DIN + s * 32 + k;
                else if (s < 8) src = agg + ((size_t)rowG * RR + 0) * DIN + (s - 4) * 32 + k;
                else            src = agg + ((size_t)rowG * RR + 1) * DIN + (s - 8) * 32 + k;
                v = *(const float4*)src;
            }
            Abf[lrow][k + 0] = (__bf16)v.x;
            Abf[lrow][k + 1] = (__bf16)v.y;
            Abf[lrow][k + 2] = (__bf16)v.z;
            Abf[lrow][k + 3] = (__bf16)v.w;
        }
        // ---- stage B (transposed): 32 k x 64 cols, 2 float4 per thread ----
        #pragma unroll
        for (int q = 0; q < 2; ++q) {
            int tt = tid + q * 256;
            int kk = tt >> 4;
            int c4 = (tt & 15) * 4;
            const float* wsrc;
            if (s < 4) {
                wsrc = w1s + (size_t)(s * 32 + kk) * DH + colBase + c4;
            } else {
                int r    = (s - 4) >> 2;
                int koff = ((s - 4) & 3) * 32;
                wsrc = w1r + (size_t)(r * DIN + koff + kk) * DH + colBase + c4;
            }
            float4 wv = *(const float4*)wsrc;
            Bbf[c4 + 0][kk] = (__bf16)wv.x;
            Bbf[c4 + 1][kk] = (__bf16)wv.y;
            Bbf[c4 + 2][kk] = (__bf16)wv.z;
            Bbf[c4 + 3][kk] = (__bf16)wv.w;
        }
        __syncthreads();

        // ---- assemble fragments per ISA 7.12.2 and issue WMMA ----
        const __bf16* arow = &Abf[wave * 16 + l16][0];
        const int k0 = half * 8;
        const int k1 = 16 + half * 8;
        v16bf afrag;
        #pragma unroll
        for (int j = 0; j < 8; ++j) {
            afrag[j]     = arow[k0 + j];
            afrag[8 + j] = arow[k1 + j];
        }
        #pragma unroll
        for (int t = 0; t < 4; ++t) {
            const __bf16* brow = &Bbf[t * 16 + l16][0];
            v16bf bfrag;
            #pragma unroll
            for (int j = 0; j < 16; ++j) bfrag[j] = brow[(j < 8) ? (half * 8 + j) : (16 + half * 8 + (j - 8))];
            acc[t] = __builtin_amdgcn_wmma_f32_16x16x32_bf16(
                false, afrag, false, bfrag, (short)0, acc[t], false, false);
        }
        __syncthreads();
    }

    // ---- epilogue: +bias, ReLU, store h ----
    #pragma unroll
    for (int t = 0; t < 4; ++t) {
        int col    = colBase + t * 16 + l16;
        float bias = b1[col];
        #pragma unroll
        for (int v = 0; v < 8; ++v) {
            int row = rowBase + wave * 16 + v + half * 8;
            if (row < NN) h[(size_t)row * DH + col] = fmaxf(acc[t][v] + bias, 0.f);
        }
    }
}

// ---------------------------------------------------------------------------
// Kernel 4: layer-2 projections (linear-before-scatter trick):
//   out[i]      = h[i] @ W2_self + b2          (written, later atomically added to)
//   zrel[i,r,:] = h[i] @ W2_rel[r]             (2 floats per (node, relation))
// ---------------------------------------------------------------------------
__global__ __launch_bounds__(256) void k_proj2(const float* __restrict__ h,
                                               const float* __restrict__ w2s,
                                               const float* __restrict__ w2r,
                                               const float* __restrict__ b2,
                                               float* __restrict__ out,
                                               float* __restrict__ zrel) {
    __shared__ float W[6][DH];   // col-major weight columns: s0,s1,r0c0,r0c1,r1c0,r1c1
    int t = threadIdx.x;
    for (int k = t; k < DH; k += 256) {
        W[0][k] = w2s[2 * k];
        W[1][k] = w2s[2 * k + 1];
        W[2][k] = w2r[2 * k];
        W[3][k] = w2r[2 * k + 1];
        W[4][k] = w2r[2 * DH + 2 * k];
        W[5][k] = w2r[2 * DH + 2 * k + 1];
    }
    __syncthreads();

    int i = blockIdx.x * 256 + t;
    if (i >= NN) return;
    const float* hr = h + (size_t)i * DH;
    float s0 = 0.f, s1 = 0.f, r00 = 0.f, r01 = 0.f, r10 = 0.f, r11 = 0.f;
    #pragma unroll 4
    for (int k = 0; k < DH; k += 4) {
        float4 hv = *(const float4*)(hr + k);
        s0  += hv.x * W[0][k] + hv.y * W[0][k+1] + hv.z * W[0][k+2] + hv.w * W[0][k+3];
        s1  += hv.x * W[1][k] + hv.y * W[1][k+1] + hv.z * W[1][k+2] + hv.w * W[1][k+3];
        r00 += hv.x * W[2][k] + hv.y * W[2][k+1] + hv.z * W[2][k+2] + hv.w * W[2][k+3];
        r01 += hv.x * W[3][k] + hv.y * W[3][k+1] + hv.z * W[3][k+2] + hv.w * W[3][k+3];
        r10 += hv.x * W[4][k] + hv.y * W[4][k+1] + hv.z * W[4][k+2] + hv.w * W[4][k+3];
        r11 += hv.x * W[5][k] + hv.y * W[5][k+1] + hv.z * W[5][k+2] + hv.w * W[5][k+3];
    }
    out[2 * i]     = s0 + b2[0];
    out[2 * i + 1] = s1 + b2[1];
    zrel[4 * i]     = r00;
    zrel[4 * i + 1] = r01;
    zrel[4 * i + 2] = r10;
    zrel[4 * i + 3] = r11;
}

// ---------------------------------------------------------------------------
// Kernel 5: layer-2 edge scatter — only 8 bytes of payload per edge.
// ---------------------------------------------------------------------------
__global__ __launch_bounds__(256) void k_scatter2(const int* __restrict__ ei,
                                                  const int* __restrict__ et,
                                                  const float* __restrict__ zrel,
                                                  float* __restrict__ out) {
    int e = blockIdx.x * 256 + threadIdx.x;
    if (e >= EE) return;
    int src = ei[e];
    int dst = ei[EE + e];
    int r   = et[e];
    const float* z = zrel + ((size_t)src * RR + r) * DOUT;
    unsafeAtomicAdd(out + 2 * (size_t)dst,     z[0]);
    unsafeAtomicAdd(out + 2 * (size_t)dst + 1, z[1]);
}

// ---------------------------------------------------------------------------
extern "C" void kernel_launch(void* const* d_in, const int* in_sizes, int n_in,
                              void* d_out, int out_size, void* d_ws, size_t ws_size,
                              hipStream_t stream) {
    const float* x   = (const float*)d_in[0];
    const int*   ei  = (const int*)d_in[1];
    const int*   et  = (const int*)d_in[2];
    const float* w1s = (const float*)d_in[3];
    const float* w1r = (const float*)d_in[4];
    const float* b1  = (const float*)d_in[5];
    const float* w2s = (const float*)d_in[6];
    const float* w2r = (const float*)d_in[7];
    const float* b2  = (const float*)d_in[8];
    float* out = (float*)d_out;

    char* ws = (char*)d_ws;
    const size_t aggBytes = (size_t)NN * RR * DIN * sizeof(float);  // 102,400,000
    const size_t hBytes   = (size_t)NN * DH * sizeof(float);        // 102,400,000
    float* agg  = (float*)ws;
    float* h    = (float*)(ws + aggBytes);
    float* zrel = (float*)(ws + aggBytes + hBytes);

    // 1) zero agg (atomics accumulate; must re-zero every call)
    {
        int n4 = (NN * RR * DIN) / 4;            // 6,400,000
        k_zero<<<(n4 + 255) / 256, 256, 0, stream>>>((float4*)agg, n4);
    }
    // 2) layer-1 scatter: 32 lanes per edge
    {
        long long threads = (long long)EE * 32;
        int blocks = (int)((threads + 255) / 256); // 200,000
        k_scatter1<<<blocks, 256, 0, stream>>>(x, ei, et, agg);
    }
    // 3) fused WMMA GEMM + bias + ReLU -> h
    {
        int rowTiles = (NN + 127) / 128;          // 782
        k_gemm1<<<rowTiles * 4, 256, 0, stream>>>(x, agg, w1s, w1r, b1, h);
    }
    // 4) layer-2 projections -> out (self part + bias) and zrel
    {
        k_proj2<<<(NN + 255) / 256, 256, 0, stream>>>(h, w2s, w2r, b2, out, zrel);
    }
    // 5) layer-2 scatter of 2-float payloads
    {
        k_scatter2<<<(EE + 255) / 256, 256, 0, stream>>>(ei, et, zrel, out);
    }
}
